// EfficientDynamicAttention_30717606101535
// MI455X (gfx1250) — matrix operand order repaired and checked by reading, not compiled, not run on hardware
//
#include <hip/hip_runtime.h>
#include <cstdint>
#include <cstddef>

#define FRAMES 32
#define NTOK   2304
#define CH     512
#define C3     1536
#define DH     64
#define NHEADS 8
#define ROWS   128
#define TT     16
#define EPSV   1e-6f

typedef __attribute__((ext_vector_type(16))) __bf16 v16bf;
typedef __attribute__((ext_vector_type(8)))  __bf16 v8bf;
typedef __attribute__((ext_vector_type(8)))  float  v8f;
typedef __attribute__((ext_vector_type(4)))  int    i4v;

#if defined(__AMDGCN__) && __has_builtin(__builtin_amdgcn_global_load_async_to_lds_b128)
#define HAVE_ASYNC_LDS 1
#endif

__device__ __forceinline__ v8f wmma_bf(v16bf a, v16bf b, v8f c) {
  return __builtin_amdgcn_wmma_f32_16x16x32_bf16(false, a, false, b, (short)0, c,
                                                 false, false);
}
__device__ __forceinline__ v16bf cat16(v8bf lo, v8bf hi) {
  v16bf r;
#pragma unroll
  for (int i = 0; i < 8; ++i) { r[i] = lo[i]; r[i + 8] = hi[i]; }
  return r;
}
__device__ __forceinline__ v8f zero8() {
  v8f r;
#pragma unroll
  for (int i = 0; i < 8; ++i) r[i] = 0.f;
  return r;
}
__device__ __forceinline__ unsigned short f2h(float f) {
  union { float f; unsigned u; } v; v.f = f;
  unsigned r = v.u + 0x7FFFu + ((v.u >> 16) & 1u);  // round-nearest-even bf16
  return (unsigned short)(r >> 16);
}
__device__ __forceinline__ float fmap(float x) {  // elu(x)+1
  return x > 0.f ? x + 1.f : __expf(x);
}

// Stage 16 contiguous bf16 (32B) global -> LDS. Uses gfx1250 async-to-LDS DMA
// (ASYNCcnt path) when available, else VGPR bounce.
__device__ __forceinline__ void stage32B(const unsigned short* g,
                                         unsigned short* l) {
#ifdef HAVE_ASYNC_LDS
  __attribute__((address_space(1))) i4v* gp =
      (__attribute__((address_space(1))) i4v*)g;
  __attribute__((address_space(3))) i4v* lp =
      (__attribute__((address_space(3))) i4v*)l;
  __builtin_amdgcn_global_load_async_to_lds_b128(gp, lp, 0, 0);
  __builtin_amdgcn_global_load_async_to_lds_b128(gp + 1, lp + 1, 0, 0);
#else
  uint4 d0 = *(const uint4*)g;
  uint4 d1 = *(const uint4*)(g + 8);
  *(uint4*)l = d0;
  *(uint4*)(l + 8) = d1;
#endif
}
__device__ __forceinline__ void async_fence() {
#ifdef HAVE_ASYNC_LDS
#if __has_builtin(__builtin_amdgcn_s_wait_asynccnt)
  __builtin_amdgcn_s_wait_asynccnt(0);
#else
  asm volatile("s_wait_asynccnt 0x0" ::: "memory");
#endif
#endif
}

// ---------------- temporal path (tiny) ----------------

__global__ void __launch_bounds__(256)
k_reduce(const float* __restrict__ x, const float* __restrict__ temb,
         const float* __restrict__ motion, float* __restrict__ xtime,
         float* __restrict__ mwt) {
  int bt = blockIdx.x, tid = threadIdx.x;
  const float* xb = x + (size_t)bt * NTOK * CH;
#pragma unroll
  for (int rep = 0; rep < 2; ++rep) {
    int c = tid + rep * 256;
    float s = 0.f;
    for (int n = 0; n < NTOK; ++n) s += xb[(size_t)n * CH + c];
    xtime[bt * CH + c] = s * (1.f / NTOK) + temb[c];
  }
  __shared__ float red[256];
  float m = 0.f;
  const float* mb = motion + (size_t)bt * NTOK;
  for (int n = tid; n < NTOK; n += 256) m += mb[n];
  red[tid] = m;
  __syncthreads();
  for (int s2 = 128; s2 > 0; s2 >>= 1) {
    if (tid < s2) red[tid] += red[tid + s2];
    __syncthreads();
  }
  if (tid == 0) mwt[bt] = 1.f + tanhf(red[0] * (1.f / NTOK));
}

__global__ void __launch_bounds__(256)
k_qkvt(const float* __restrict__ xtime, const float* __restrict__ W,
       float* __restrict__ qkvt) {
  int bt = blockIdx.x;
  __shared__ float row[CH];
  for (int i = threadIdx.x; i < CH; i += 256) row[i] = xtime[bt * CH + i];
  __syncthreads();
  for (int col = threadIdx.x; col < C3; col += 256) {
    float s = 0.f;
    for (int k = 0; k < CH; ++k) s += row[k] * W[(size_t)k * C3 + col];
    qkvt[(size_t)bt * C3 + col] = s;
  }
}

__global__ void __launch_bounds__(256)
k_tattn(const float* __restrict__ qkvt, const float* __restrict__ mwt,
        float* __restrict__ tout) {
  int b = blockIdx.x >> 3, h = blockIdx.x & 7, tid = threadIdx.x;
  __shared__ float sq[TT * DH], sk[TT * DH], sv[TT * DH];
  __shared__ float skv[DH * DH], sks[DH], szd[TT];
  for (int i = tid; i < TT * DH; i += 256) {
    int t = i >> 6, d = i & 63;
    size_t base = (size_t)(b * TT + t) * C3;
    sq[i] = fmap(qkvt[base + h * DH + d]) * mwt[b * TT + t];
    sk[i] = fmap(qkvt[base + CH + h * DH + d]);
    sv[i] = qkvt[base + 2 * CH + h * DH + d];
  }
  __syncthreads();
  for (int i = tid; i < DH * DH; i += 256) {
    int d = i >> 6, e = i & 63;
    float s = 0.f;
#pragma unroll
    for (int t = 0; t < TT; ++t) s += sk[t * DH + d] * sv[t * DH + e];
    skv[i] = s;
  }
  if (tid < DH) {
    float s = 0.f;
#pragma unroll
    for (int t = 0; t < TT; ++t) s += sk[t * DH + tid];
    sks[tid] = s;
  }
  __syncthreads();
  if (tid < TT) {
    float s = 0.f;
    for (int d = 0; d < DH; ++d) s += sq[tid * DH + d] * sks[d];
    szd[tid] = 1.f / (s + EPSV);
  }
  __syncthreads();
  for (int i = tid; i < TT * DH; i += 256) {
    int t = i >> 6, e = i & 63;
    float s = 0.f;
    for (int d = 0; d < DH; ++d) s += sq[t * DH + d] * skv[d * DH + e];
    tout[(size_t)(b * TT + t) * CH + h * DH + e] = s * szd[t];
  }
}

// ---------------- bf16 prep ----------------

__global__ void __launch_bounds__(256)
k_wcvt(const float* __restrict__ W, unsigned short* __restrict__ Wbf) {
  int i = blockIdx.x * 256 + threadIdx.x;
  if (i < CH * C3) Wbf[i] = f2h(W[i]);
}

__global__ void __launch_bounds__(256)
k_prep(const float* __restrict__ x, const float* __restrict__ temb,
       const float* __restrict__ tout, unsigned short* __restrict__ xsp) {
  size_t i = ((size_t)blockIdx.x * 256 + threadIdx.x) * 4;
  const size_t total = (size_t)FRAMES * NTOK * CH;
  if (i >= total) return;
  float4 v = *(const float4*)(x + i);
  int c = (int)(i & (CH - 1));
  size_t bt = i / ((size_t)NTOK * CH);
  float4 te = *(const float4*)(temb + c);
  float4 to = *(const float4*)(tout + bt * CH + c);
  uint2 p;
  p.x = (unsigned)f2h(v.x + te.x + to.x) |
        ((unsigned)f2h(v.y + te.y + to.y) << 16);
  p.y = (unsigned)f2h(v.z + te.z + to.z) |
        ((unsigned)f2h(v.w + te.w + to.w) << 16);
  *(uint2*)(xsp + i) = p;
}

__global__ void __launch_bounds__(256)
k_zero(float* __restrict__ p, size_t n) {
  size_t i = (size_t)blockIdx.x * 256 + threadIdx.x;
  if (i < n) p[i] = 0.f;
}

// ---------------- pass A: K,V GEMM + kv = K^T V, k_sum ----------------

__global__ void __launch_bounds__(256)
k_passA(const unsigned short* __restrict__ xsp,
        const unsigned short* __restrict__ Wbf,
        float* __restrict__ kvbuf, float* __restrict__ ksum) {
  __shared__ unsigned short smem[16384];  // 32 KB
  unsigned short* sA = smem;              // [128][32] row-major (row, k)
  unsigned short* sB = smem + 4096;       // [128][32] col-major-in-lds (n, k)
  int bt = blockIdx.x, tile = blockIdx.y, h = blockIdx.z;
  int tid = threadIdx.x, lane = tid & 31, wave = tid >> 5;
  int hf = lane >> 4, l15 = lane & 15;
  int rowbase = tile * ROWS;

  v8f ak[4], av[4];
#pragma unroll
  for (int i = 0; i < 4; ++i) { ak[i] = zero8(); av[i] = zero8(); }

  const size_t xbase = ((size_t)bt * NTOK + rowbase) * CH;
  const int rA = tid >> 1, kkA = (tid & 1) << 4;
  for (int kb = 0; kb < CH; kb += 32) {
    __syncthreads();
    {  // stage A tile: 128 x 32 bf16 via async-to-LDS DMA
      const unsigned short* src = xsp + xbase + (size_t)rA * CH + kb + kkA;
      stage32B(src, &sA[rA * 32 + kkA]);
      if (kb + 32 < CH) __builtin_prefetch(src + 32, 0, 3);
    }
    {  // stage B (W columns for this head: 64 K-cols then 64 V-cols), transposed
      int k = tid >> 3, seg = tid & 7;
      int col = (seg < 4) ? (CH + h * DH + seg * 16)
                          : (2 * CH + h * DH + (seg - 4) * 16);
      const unsigned short* src = Wbf + (size_t)(kb + k) * C3 + col;
#pragma unroll
      for (int j = 0; j < 16; ++j) sB[(seg * 16 + j) * 32 + k] = src[j];
    }
    async_fence();
    __syncthreads();
    v16bf afrag = cat16(*(const v8bf*)&sA[(wave * 16 + l15) * 32 + hf * 16],
                        *(const v8bf*)&sA[(wave * 16 + l15) * 32 + hf * 16 + 8]);
    v16bf bk[4], bv[4];
#pragma unroll
    for (int c = 0; c < 4; ++c) {  // load all B fragments first
      int n = c * 16 + l15;
      bk[c] = cat16(*(const v8bf*)&sB[n * 32 + hf * 16],
                    *(const v8bf*)&sB[n * 32 + hf * 16 + 8]);
      bv[c] = cat16(*(const v8bf*)&sB[(64 + n) * 32 + hf * 16],
                    *(const v8bf*)&sB[(64 + n) * 32 + hf * 16 + 8]);
    }
#pragma unroll
    for (int c = 0; c < 4; ++c) ak[c] = wmma_bf(afrag, bk[c], ak[c]);
#pragma unroll
    for (int c = 0; c < 4; ++c) av[c] = wmma_bf(afrag, bv[c], av[c]);
  }

  __syncthreads();  // done reading sA/sB; repurpose LDS as kT/vT
  unsigned short* kT = smem;         // [64(d)][128(row)]
  unsigned short* vT = smem + 8192;  // [64(e)][128(row)]
  float csum[4];
#pragma unroll
  for (int c = 0; c < 4; ++c) {
    float cs = 0.f;
#pragma unroll
    for (int j = 0; j < 8; ++j) {
      float kval = fmap(ak[c][j]);
      cs += kval;
      int row = wave * 16 + j + 8 * hf;
      int d = c * 16 + l15;
      kT[d * 128 + row] = f2h(kval);
      vT[d * 128 + row] = f2h(av[c][j]);
    }
    cs += __shfl_xor(cs, 16);
    csum[c] = cs;
  }
  if (hf == 0) {
#pragma unroll
    for (int c = 0; c < 4; ++c)
      atomicAdd(&ksum[((size_t)bt * NHEADS + h) * DH + c * 16 + l15], csum[c]);
  }
  __syncthreads();

  // kv = K^T V : M=64(d) K=128(rows) N=64(e) -> 16 tiles, 2 per wave
  float* kvdst = kvbuf + ((size_t)bt * NHEADS + h) * DH * DH;
#pragma unroll
  for (int ti = 0; ti < 2; ++ti) {
    int t = wave * 2 + ti;
    int dt = t >> 2, et = t & 3;
    v8f acc = zero8();
#pragma unroll
    for (int r0 = 0; r0 < 128; r0 += 32) {
      v16bf a = cat16(*(const v8bf*)&kT[(dt * 16 + l15) * 128 + r0 + hf * 16],
                      *(const v8bf*)&kT[(dt * 16 + l15) * 128 + r0 + hf * 16 + 8]);
      v16bf b = cat16(*(const v8bf*)&vT[(et * 16 + l15) * 128 + r0 + hf * 16],
                      *(const v8bf*)&vT[(et * 16 + l15) * 128 + r0 + hf * 16 + 8]);
      acc = wmma_bf(a, b, acc);
    }
#pragma unroll
    for (int j = 0; j < 8; ++j) {
      int d = dt * 16 + j + 8 * hf;
      int e = et * 16 + l15;
      atomicAdd(&kvdst[d * DH + e], acc[j]);
    }
  }
}

// ---------------- pass B: Q GEMM + out = Q kv, z normalize ----------------

__global__ void __launch_bounds__(256)
k_passB(const unsigned short* __restrict__ xsp,
        const unsigned short* __restrict__ Wbf,
        const float* __restrict__ kvbuf, const float* __restrict__ ksum,
        const float* __restrict__ motion, float* __restrict__ out) {
  __shared__ unsigned short smem[12288];  // 24 KB
  __shared__ float sKS[DH];
  unsigned short* sA = smem;              // [128][32]
  unsigned short* sB = smem + 4096;       // [64][32] (n, k)
  int bt = blockIdx.x, tile = blockIdx.y, h = blockIdx.z;
  int tid = threadIdx.x, lane = tid & 31, wave = tid >> 5;
  int hf = lane >> 4, l15 = lane & 15;
  int rowbase = tile * ROWS;

  if (tid < DH) sKS[tid] = ksum[((size_t)bt * NHEADS + h) * DH + tid];

  v8f aq[4];
#pragma unroll
  for (int i = 0; i < 4; ++i) aq[i] = zero8();

  const size_t xbase = ((size_t)bt * NTOK + rowbase) * CH;
  const int rA = tid >> 1, kkA = (tid & 1) << 4;
  for (int kb = 0; kb < CH; kb += 32) {
    __syncthreads();
    {
      const unsigned short* src = xsp + xbase + (size_t)rA * CH + kb + kkA;
      stage32B(src, &sA[rA * 32 + kkA]);
      if (kb + 32 < CH) __builtin_prefetch(src + 32, 0, 3);
    }
    if (tid < 128) {  // stage W q-columns, transposed
      int k = tid >> 2, seg = tid & 3;
      const unsigned short* src = Wbf + (size_t)(kb + k) * C3 + h * DH + seg * 16;
#pragma unroll
      for (int j = 0; j < 16; ++j) sB[(seg * 16 + j) * 32 + k] = src[j];
    }
    async_fence();
    __syncthreads();
    v16bf afrag = cat16(*(const v8bf*)&sA[(wave * 16 + l15) * 32 + hf * 16],
                        *(const v8bf*)&sA[(wave * 16 + l15) * 32 + hf * 16 + 8]);
    v16bf bq[4];
#pragma unroll
    for (int c = 0; c < 4; ++c) {
      int n = c * 16 + l15;
      bq[c] = cat16(*(const v8bf*)&sB[n * 32 + hf * 16],
                    *(const v8bf*)&sB[n * 32 + hf * 16 + 8]);
    }
#pragma unroll
    for (int c = 0; c < 4; ++c) aq[c] = wmma_bf(afrag, bq[c], aq[c]);
  }

  float mwr[8], zden[8];
#pragma unroll
  for (int j = 0; j < 8; ++j) {
    int row = rowbase + wave * 16 + j + 8 * hf;
    mwr[j] = 1.f + tanhf(motion[(size_t)bt * NTOK + row]);
    zden[j] = 0.f;
  }
  __syncthreads();  // repurpose LDS
  unsigned short* qlds = smem;         // 8 waves x [16(row)][64(d)]
  unsigned short* kvb = smem + 8192;   // [64(e)][64(d)]
#pragma unroll
  for (int c = 0; c < 4; ++c) {
#pragma unroll
    for (int j = 0; j < 8; ++j) {
      float q = fmap(aq[c][j]) * mwr[j];
      zden[j] += q * sKS[c * 16 + l15];
      qlds[wave * 1024 + (j + 8 * hf) * 64 + c * 16 + l15] = f2h(q);
    }
  }
#pragma unroll
  for (int j = 0; j < 8; ++j) {
    float d = zden[j];
    d += __shfl_xor(d, 1); d += __shfl_xor(d, 2);
    d += __shfl_xor(d, 4); d += __shfl_xor(d, 8);
    zden[j] = 1.f / (d + EPSV);
  }
  {
    const float* kvsrc = kvbuf + ((size_t)bt * NHEADS + h) * DH * DH;
    for (int i = tid; i < DH * DH; i += 256) {
      int d = i >> 6, e = i & 63;
      kvb[e * 64 + d] = f2h(kvsrc[i]);  // transposed for B-fragment reads
    }
  }
  __syncthreads();

  v8f ao[4];
#pragma unroll
  for (int i = 0; i < 4; ++i) ao[i] = zero8();
#pragma unroll
  for (int kq = 0; kq < 2; ++kq) {
    v16bf a =
        cat16(*(const v8bf*)&qlds[wave * 1024 + l15 * 64 + kq * 32 + hf * 16],
              *(const v8bf*)&qlds[wave * 1024 + l15 * 64 + kq * 32 + hf * 16 + 8]);
#pragma unroll
    for (int c = 0; c < 4; ++c) {
      v16bf b = cat16(*(const v8bf*)&kvb[(c * 16 + l15) * 64 + kq * 32 + hf * 16],
                      *(const v8bf*)&kvb[(c * 16 + l15) * 64 + kq * 32 + hf * 16 + 8]);
      ao[c] = wmma_bf(a, b, ao[c]);
    }
  }
#pragma unroll
  for (int c = 0; c < 4; ++c) {
#pragma unroll
    for (int j = 0; j < 8; ++j) {
      int row = rowbase + wave * 16 + j + 8 * hf;
      out[((size_t)bt * NTOK + row) * CH + h * DH + c * 16 + l15] =
          ao[c][j] * zden[j];
    }
  }
}

// ---------------- launch ----------------

extern "C" void kernel_launch(void* const* d_in, const int* in_sizes, int n_in,
                              void* d_out, int out_size, void* d_ws,
                              size_t ws_size, hipStream_t stream) {
  const float* x = (const float*)d_in[0];
  const float* motion = (const float*)d_in[1];
  const float* W = (const float*)d_in[2];
  const float* temb = (const float*)d_in[3];
  float* out = (float*)d_out;

  char* ws = (char*)d_ws;
  size_t off = 0;
  auto alloc = [&](size_t bytes) -> void* {
    void* p = ws + off;
    off = (off + bytes + 255) & ~(size_t)255;
    return p;
  };
  unsigned short* Wbf = (unsigned short*)alloc((size_t)CH * C3 * 2);
  float* xtime = (float*)alloc((size_t)FRAMES * CH * 4);
  float* qkvt = (float*)alloc((size_t)FRAMES * C3 * 4);
  float* mwt = (float*)alloc((size_t)FRAMES * 4);
  float* tout = (float*)alloc((size_t)FRAMES * CH * 4);
  float* kvbuf = (float*)alloc((size_t)FRAMES * NHEADS * DH * DH * 4);
  float* ksum = (float*)alloc((size_t)FRAMES * NHEADS * DH * 4);
  unsigned short* xsp = (unsigned short*)alloc((size_t)FRAMES * NTOK * CH * 2);

  // temporal branch
  k_reduce<<<FRAMES, 256, 0, stream>>>(x, temb, motion, xtime, mwt);
  k_qkvt<<<FRAMES, 256, 0, stream>>>(xtime, W, qkvt);
  k_tattn<<<2 * NHEADS, 256, 0, stream>>>(qkvt, mwt, tout);

  // bf16 prep
  k_wcvt<<<(CH * C3 + 255) / 256, 256, 0, stream>>>(W, Wbf);
  {
    size_t total = (size_t)FRAMES * NTOK * CH;
    k_prep<<<(unsigned)((total / 4 + 255) / 256), 256, 0, stream>>>(x, temb,
                                                                   tout, xsp);
  }
  {
    size_t n = (size_t)FRAMES * NHEADS * DH * DH + (size_t)FRAMES * NHEADS * DH;
    k_zero<<<(unsigned)((n + 255) / 256), 256, 0, stream>>>(kvbuf, n);
  }

  dim3 grid(FRAMES, NTOK / ROWS, NHEADS);
  k_passA<<<grid, 256, 0, stream>>>(xsp, Wbf, kvbuf, ksum);
  k_passB<<<grid, 256, 0, stream>>>(xsp, Wbf, kvbuf, ksum, motion, out);
}
